// iekt_55929064128830
// MI455X (gfx1250) — compile-verified, matrix operand order checked by reading, not run on hardware
//
#include <hip/hip_runtime.h>
#include <math.h>

typedef _Float16 h16;
typedef __attribute__((ext_vector_type(8)))  _Float16 v8h;
typedef __attribute__((ext_vector_type(16))) _Float16 v16h;
typedef __attribute__((ext_vector_type(8)))  float    v8f;

#define B_SZ  16384
#define D_SZ  64
#define MAXC  8
#define NPROB 50000
#define NCON  2000
#define TDIM  1024
#define HID   256

// ---------------------------------------------------------------------------
// Weight transpose + fp32 -> f16 convert: src [K x N] row-major -> dst [N x K]
// ---------------------------------------------------------------------------
__global__ void transpose_f32_to_f16_kernel(const float* __restrict__ src,
                                            h16* __restrict__ dst, int K, int N) {
  int idx = blockIdx.x * blockDim.x + threadIdx.x;
  if (idx >= K * N) return;
  int k = idx / N, n = idx - k * N;
  dst[(size_t)n * K + k] = (h16)src[idx];
}

// Gather rows by index and convert to f16 (fuses red_prob[prob_ids] input).
__global__ void gather_convert_kernel(const float* __restrict__ table,
                                      const int* __restrict__ ids,
                                      h16* __restrict__ dst, int cols) {
  int row = blockIdx.x;
  const float* s = table + (size_t)ids[row] * cols;
  h16* d = dst + (size_t)row * cols;
  for (int c = threadIdx.x; c < cols; c += blockDim.x) d[c] = (h16)s[c];
}

// Convert concept table to f16, zero-padding rows [srcRows, gridDim.x).
__global__ void convert_pad_kernel(const float* __restrict__ src,
                                   h16* __restrict__ dst, int srcRows, int cols) {
  int row = blockIdx.x;
  h16* d = dst + (size_t)row * cols;
  if (row < srcRows) {
    const float* s = src + (size_t)row * cols;
    for (int c = threadIdx.x; c < cols; c += blockDim.x) d[c] = (h16)s[c];
  } else {
    for (int c = threadIdx.x; c < cols; c += blockDim.x) d[c] = (h16)0.f;
  }
}

// ---------------------------------------------------------------------------
// WMMA GEMM: C[M x N] = act(A[M x K] @ B[K x N] + bias), Bt is B transposed
// [N x K] f16 row-major. Block = 128 threads = 4 waves; block tile 64x64.
// Wave w -> rows [blockIdx.x*64 + w*16), 4 accumulators across 64 columns.
// Fragment layout per CDNA5 ISA (16-bit A 16x32): element e of the
// v16h holds K = k0 + (e>>3)*16 + (lane>>4)*8 + (e&7)  -> two 16B loads.
// ---------------------------------------------------------------------------
__global__ __launch_bounds__(128)
void gemm_wmma_kernel(const h16* __restrict__ A, const h16* __restrict__ Bt,
                      const float* __restrict__ bias,
                      float* __restrict__ Cf, h16* __restrict__ Ch,
                      int M, int N, int K, int relu) {
  int wave = threadIdx.x >> 5;
  int lane = threadIdx.x & 31;
  int hf   = lane >> 4;      // K-half selector for A/B fragments
  int l16  = lane & 15;      // row (A) / column (B,C) within tile
  int mWave = blockIdx.x * 64 + wave * 16;
  int nBase = blockIdx.y * 64;

  const h16* Arow  = A  + (size_t)(mWave + l16) * K;
  const h16* Brow0 = Bt + (size_t)(nBase + l16) * K;

  v8f acc0 = {}, acc1 = {}, acc2 = {}, acc3 = {};

  for (int k0 = 0; k0 < K; k0 += 32) {
    v8h alo = *(const v8h*)(Arow + k0 + hf * 8);
    v8h ahi = *(const v8h*)(Arow + k0 + 16 + hf * 8);
    v16h af = __builtin_shufflevector(alo, ahi,
                0,1,2,3,4,5,6,7,8,9,10,11,12,13,14,15);
#define BSTEP(i)                                                               \
    {                                                                          \
      const h16* br = Brow0 + (size_t)(i) * 16 * K;                            \
      v8h blo = *(const v8h*)(br + k0 + hf * 8);                               \
      v8h bhi = *(const v8h*)(br + k0 + 16 + hf * 8);                          \
      v16h bf = __builtin_shufflevector(blo, bhi,                              \
                  0,1,2,3,4,5,6,7,8,9,10,11,12,13,14,15);                      \
      acc##i = __builtin_amdgcn_wmma_f32_16x16x32_f16(                         \
          false, af, false, bf, (short)0, acc##i, false, false);               \
    }
    BSTEP(0) BSTEP(1) BSTEP(2) BSTEP(3)
#undef BSTEP
  }

  // C layout: lane half selects M block (half*8 + r), l16 = column.
#define EPI(i)                                                                 \
  {                                                                            \
    int col = nBase + (i) * 16 + l16;                                          \
    float bv = bias ? bias[col] : 0.f;                                         \
    _Pragma("unroll")                                                          \
    for (int r = 0; r < 8; ++r) {                                              \
      int row = mWave + hf * 8 + r;                                            \
      float val = acc##i[r] + bv;                                              \
      if (relu) val = fmaxf(val, 0.f);                                         \
      if (Cf) Cf[(size_t)row * N + col] = val;                                 \
      if (Ch) Ch[(size_t)row * N + col] = (h16)val;                            \
    }                                                                          \
  }
  EPI(0) EPI(1) EPI(2) EPI(3)
#undef EPI
}

// ---------------------------------------------------------------------------
// Cosine similarity (raw fp32 embeddings) + mask + softmax + weighted concept
// sum; builds v = [red_prob_row(64), rep(64)]. One block (8 waves) per row.
// ---------------------------------------------------------------------------
__global__ __launch_bounds__(256)
void cos_softmax_rep_kernel(const float* __restrict__ ex, const float* __restrict__ con,
                            const int* __restrict__ prob_ids, const int* __restrict__ rel,
                            const int* __restrict__ filt,
                            const float* __restrict__ redQ, const float* __restrict__ redC,
                            float* __restrict__ v) {
  int b = blockIdx.x;
  int m = threadIdx.x >> 5;        // concept slot 0..7 (one wave each)
  int lane = threadIdx.x & 31;
  __shared__ float tw_s[MAXC];

  int pid = prob_ids[b];
  int cid = rel[b * MAXC + m];
  const float* q = ex + (size_t)pid * TDIM;
  const float* c = con + (size_t)cid * TDIM;

  float dot = 0.f, qq = 0.f, cc = 0.f;
  for (int i = lane; i < TDIM; i += 32) {
    float qv = q[i], cv = c[i];
    dot += qv * cv; qq += qv * qv; cc += cv * cv;
  }
  for (int off = 16; off > 0; off >>= 1) {
    dot += __shfl_xor(dot, off, 32);
    qq  += __shfl_xor(qq,  off, 32);
    cc  += __shfl_xor(cc,  off, 32);
  }
  if (lane == 0) {
    float t = dot / fmaxf(sqrtf(qq) * sqrtf(cc), 1e-8f);
    if (filt[b * MAXC + m] == 0) t = -1e9f;
    tw_s[m] = t;
  }
  __syncthreads();

  if (threadIdx.x < 64) {
    int d = threadIdx.x;
    float mx = tw_s[0];
    #pragma unroll
    for (int j = 1; j < MAXC; ++j) mx = fmaxf(mx, tw_s[j]);
    float w[MAXC]; float s = 0.f;
    #pragma unroll
    for (int j = 0; j < MAXC; ++j) { w[j] = __expf(tw_s[j] - mx); s += w[j]; }
    float inv = 1.f / s;
    float rep = 0.f;
    #pragma unroll
    for (int j = 0; j < MAXC; ++j)
      rep += (w[j] * inv) * redC[(size_t)rel[b * MAXC + j] * D_SZ + d];
    v[(size_t)b * 128 + 64 + d] = rep;
    v[(size_t)b * 128 + d]      = redQ[(size_t)b * D_SZ + d];
  }
}

// ---------------------------------------------------------------------------
// Build GRU input x = [v*op + emb*(1-op), v*(1-op) + emb*op] in f16; h -> f16.
// ---------------------------------------------------------------------------
__global__ void build_x_kernel(const float* __restrict__ v, const float* __restrict__ emb,
                               const float* __restrict__ operate, const float* __restrict__ h,
                               h16* __restrict__ xh, h16* __restrict__ hh) {
  int t = blockIdx.x * blockDim.x + threadIdx.x;
  if (t >= B_SZ * 128) return;
  int b = t >> 7, i = t & 127;
  float op = operate[b];
  float vv = v[(size_t)b * 128 + i];
  float ee = emb[(size_t)b * 128 + i];
  xh[(size_t)b * 256 + i]       = (h16)(vv * op + ee * (1.f - op));
  xh[(size_t)b * 256 + 128 + i] = (h16)(vv * (1.f - op) + ee * op);
  if (i < 64) hh[(size_t)b * 64 + i] = (h16)h[(size_t)b * 64 + i];
}

// ---------------------------------------------------------------------------
// GRU gates + h_next + predictor dot-product. One 64-thread block per row.
// ---------------------------------------------------------------------------
__global__ __launch_bounds__(64)
void finalize_kernel(const float* __restrict__ xg, const float* __restrict__ hg,
                     const float* __restrict__ h, const float* __restrict__ v,
                     const float* __restrict__ emb, const float* __restrict__ Wp,
                     const float* __restrict__ bp, float* __restrict__ out) {
  int b = blockIdx.x;
  int d = threadIdx.x;
  float xr = xg[(size_t)b * 192 + d];
  float xz = xg[(size_t)b * 192 + 64 + d];
  float xn = xg[(size_t)b * 192 + 128 + d];
  float hr = hg[(size_t)b * 192 + d];
  float hz = hg[(size_t)b * 192 + 64 + d];
  float hn = hg[(size_t)b * 192 + 128 + d];
  float r = 1.f / (1.f + __expf(-(xr + hr)));
  float z = 1.f / (1.f + __expf(-(xz + hz)));
  float n = tanhf(xn + r * hn);
  float hv = h[(size_t)b * 64 + d];
  out[(size_t)B_SZ + (size_t)b * 64 + d] = (1.f - z) * n + z * hv;

  // prob = [h(64), v(128), emb(128)] . Wp + bp
  float vd  = v[(size_t)b * 128 + d],  vd2 = v[(size_t)b * 128 + 64 + d];
  float ed  = emb[(size_t)b * 128 + d], ed2 = emb[(size_t)b * 128 + 64 + d];
  float acc = hv * Wp[d] + vd * Wp[64 + d] + vd2 * Wp[128 + d]
            + ed * Wp[192 + d] + ed2 * Wp[256 + d];
  for (int off = 16; off > 0; off >>= 1) acc += __shfl_xor(acc, off, 32);
  __shared__ float partial[2];
  if ((threadIdx.x & 31) == 0) partial[threadIdx.x >> 5] = acc;
  __syncthreads();
  if (threadIdx.x == 0) out[b] = partial[0] + partial[1] + bp[0];
}

// ---------------------------------------------------------------------------
extern "C" void kernel_launch(void* const* d_in, const int* in_sizes, int n_in,
                              void* d_out, int out_size, void* d_ws, size_t ws_size,
                              hipStream_t stream) {
  (void)in_sizes; (void)n_in; (void)out_size; (void)ws_size;
  const int*   prob_ids = (const int*)d_in[0];
  const int*   rel      = (const int*)d_in[1];
  const int*   filt     = (const int*)d_in[2];
  const float* h        = (const float*)d_in[3];
  const float* emb      = (const float*)d_in[4];
  const float* operate  = (const float*)d_in[5];
  const float* ex       = (const float*)d_in[6];
  const float* con      = (const float*)d_in[7];
  const float* W1       = (const float*)d_in[8];
  const float* b1       = (const float*)d_in[9];
  const float* W2       = (const float*)d_in[10];
  const float* b2       = (const float*)d_in[11];
  const float* Wp       = (const float*)d_in[12];
  const float* bp       = (const float*)d_in[13];
  const float* Wx       = (const float*)d_in[14];
  const float* Wh       = (const float*)d_in[15];
  const float* bx       = (const float*)d_in[16];
  const float* bh       = (const float*)d_in[17];
  float* out = (float*)d_out;

  const int NCON_P = 2048;  // concept rows padded to tile multiple

  char* ws = (char*)d_ws;
  size_t off = 0;
  auto alloc = [&](size_t bytes) -> void* {
    void* p = ws + off;
    off = (off + bytes + 255) & ~(size_t)255;
    return p;
  };
  h16*   W1t  = (h16*)  alloc((size_t)TDIM * HID * 2);     // [256][1024]
  h16*   W2t  = (h16*)  alloc((size_t)HID * D_SZ * 2);     // [64][256]
  h16*   Wxt  = (h16*)  alloc((size_t)256 * 192 * 2);      // [192][256]
  h16*   Wht  = (h16*)  alloc((size_t)64 * 192 * 2);       // [192][64]
  h16*   qh   = (h16*)  alloc((size_t)B_SZ * TDIM * 2);
  h16*   ch   = (h16*)  alloc((size_t)NCON_P * TDIM * 2);
  h16*   hidQ = (h16*)  alloc((size_t)B_SZ * HID * 2);
  h16*   hidC = (h16*)  alloc((size_t)NCON_P * HID * 2);
  float* redQ = (float*)alloc((size_t)B_SZ * D_SZ * 4);
  float* redC = (float*)alloc((size_t)NCON_P * D_SZ * 4);
  float* vbuf = (float*)alloc((size_t)B_SZ * 128 * 4);
  h16*   xh   = (h16*)  alloc((size_t)B_SZ * 256 * 2);
  h16*   hh   = (h16*)  alloc((size_t)B_SZ * 64 * 2);
  float* xg   = (float*)alloc((size_t)B_SZ * 192 * 4);
  float* hg   = (float*)alloc((size_t)B_SZ * 192 * 4);

  // Weight prep (transpose + f16 convert)
  transpose_f32_to_f16_kernel<<<(TDIM * HID + 255) / 256, 256, 0, stream>>>(W1, W1t, TDIM, HID);
  transpose_f32_to_f16_kernel<<<(HID * D_SZ + 255) / 256, 256, 0, stream>>>(W2, W2t, HID, D_SZ);
  transpose_f32_to_f16_kernel<<<(256 * 192 + 255) / 256, 256, 0, stream>>>(Wx, Wxt, 256, 192);
  transpose_f32_to_f16_kernel<<<(64 * 192 + 255) / 256, 256, 0, stream>>>(Wh, Wht, 64, 192);

  // Activation prep: gather problem rows (fuses red_prob[prob_ids]), concepts
  gather_convert_kernel<<<B_SZ, 256, 0, stream>>>(ex, prob_ids, qh, TDIM);
  convert_pad_kernel<<<NCON_P, 256, 0, stream>>>(con, ch, NCON, TDIM);

  // text_proj layer 1 (ReLU) and layer 2, for gathered problems and concepts
  gemm_wmma_kernel<<<dim3(B_SZ / 64, HID / 64), 128, 0, stream>>>(
      qh, W1t, b1, nullptr, hidQ, B_SZ, HID, TDIM, 1);
  gemm_wmma_kernel<<<dim3(NCON_P / 64, HID / 64), 128, 0, stream>>>(
      ch, W1t, b1, nullptr, hidC, NCON_P, HID, TDIM, 1);
  gemm_wmma_kernel<<<dim3(B_SZ / 64, 1), 128, 0, stream>>>(
      hidQ, W2t, b2, redQ, nullptr, B_SZ, D_SZ, HID, 0);
  gemm_wmma_kernel<<<dim3(NCON_P / 64, 1), 128, 0, stream>>>(
      hidC, W2t, b2, redC, nullptr, NCON_P, D_SZ, HID, 0);

  // cosine attention over concepts, build v
  cos_softmax_rep_kernel<<<B_SZ, 256, 0, stream>>>(ex, con, prob_ids, rel, filt,
                                                   redQ, redC, vbuf);

  // GRU input, gate GEMMs, gates + predictor
  build_x_kernel<<<(B_SZ * 128 + 255) / 256, 256, 0, stream>>>(vbuf, emb, operate, h, xh, hh);
  gemm_wmma_kernel<<<dim3(B_SZ / 64, 192 / 64), 128, 0, stream>>>(
      xh, Wxt, bx, xg, nullptr, B_SZ, 192, 256, 0);
  gemm_wmma_kernel<<<dim3(B_SZ / 64, 192 / 64), 128, 0, stream>>>(
      hh, Wht, bh, hg, nullptr, B_SZ, 192, 64, 0);
  finalize_kernel<<<B_SZ, 64, 0, stream>>>(xg, hg, h, vbuf, emb, Wp, bp, out);
}
